// HardLSTM_22557168238892
// MI455X (gfx1250) — matrix operand order repaired and checked by reading, not compiled, not run on hardware
//
#include <hip/hip_runtime.h>
#include <stdint.h>

// Problem constants (match reference)
#define TT   512
#define BB   32
#define INSZ 320
#define HH   1024
#define G4H  4096
#define NL   3
#define SLOPE 0.2f
#define OFFS  0.5f

#define NBLK 32           // persistent blocks per direction in the scan
#define THREADS 256       // 8 waves (wave32)

typedef unsigned short u16;
typedef __attribute__((ext_vector_type(16))) __bf16 bfx16;
typedef __attribute__((ext_vector_type(8)))  __bf16 bfx8;
typedef __attribute__((ext_vector_type(8)))  float  fx8;
typedef __attribute__((ext_vector_type(4)))  unsigned int ui32x4;
typedef __attribute__((ext_vector_type(8)))  int          si32x8;
typedef __attribute__((ext_vector_type(4)))  int          si32x4;

// ---------- scalar bf16 helpers (round-to-nearest-even) ----------
__device__ __forceinline__ u16 f2bf(float f) {
  unsigned x = __float_as_uint(f);
  unsigned r = (x + 0x7FFFu + ((x >> 16) & 1u)) >> 16;
  return (u16)r;
}
__device__ __forceinline__ float bf2f(u16 u) {
  return __uint_as_float((unsigned)u << 16);
}
__device__ __forceinline__ float clampf(float v, float lo, float hi) {
  return fminf(fmaxf(v, lo), hi);
}

// ---------- WMMA fragment loaders (16x16x32 bf16, wave32) ----------
// A-matrix 16x32 (M x K), row-major source, row = lane&15:
//   VGPRs 0-3: K = (lane>=16 ? 8 : 0) + 0..7
//   VGPRs 4-7: K = 16 + (lane>=16 ? 8 : 0) + 0..7
// -> two contiguous 8-element (16B) runs per lane.
__device__ __forceinline__ bfx16 load_a_frag(const u16* row, int k0, int lhi) {
  bfx8 a0 = *reinterpret_cast<const bfx8*>(row + k0 + lhi * 8);
  bfx8 a1 = *reinterpret_cast<const bfx8*>(row + k0 + 16 + lhi * 8);
  return __builtin_shufflevector(a0, a1, 0,1,2,3,4,5,6,7,8,9,10,11,12,13,14,15);
}
// B-matrix 32x16 (K x N), B = W^T with W row-major [N][K], n = lane&15:
//   VGPRs 0-3: K = (lane>=16 ? 8 : 0) + 0..7
//   VGPRs 4-7: K = 16 + (lane>=16 ? 8 : 0) + 0..7
// -> two contiguous 8-element (16B) runs per lane (same order as A).
__device__ __forceinline__ bfx16 load_b_frag(const u16* row, int k0, int lhi) {
  bfx8 b0 = *reinterpret_cast<const bfx8*>(row + k0 + lhi * 8);
  bfx8 b1 = *reinterpret_cast<const bfx8*>(row + k0 + 16 + lhi * 8);
  return __builtin_shufflevector(b0, b1, 0,1,2,3,4,5,6,7,8,9,10,11,12,13,14,15);
}

// ---------- device-wide generation barrier (per direction) ----------
__device__ __forceinline__ void gbar(unsigned* cnt, unsigned* gen, unsigned nb) {
  __threadfence();
  __syncthreads();
  if (threadIdx.x == 0) {
    unsigned g = __hip_atomic_load(gen, __ATOMIC_RELAXED, __HIP_MEMORY_SCOPE_AGENT);
    unsigned a = __hip_atomic_fetch_add(cnt, 1u, __ATOMIC_ACQ_REL, __HIP_MEMORY_SCOPE_AGENT);
    if (a == nb - 1u) {
      __hip_atomic_store(cnt, 0u, __ATOMIC_RELAXED, __HIP_MEMORY_SCOPE_AGENT);
      __hip_atomic_fetch_add(gen, 1u, __ATOMIC_RELEASE, __HIP_MEMORY_SCOPE_AGENT);
    } else {
      while (__hip_atomic_load(gen, __ATOMIC_ACQUIRE, __HIP_MEMORY_SCOPE_AGENT) == g) {
        __builtin_amdgcn_s_sleep(4);
      }
    }
  }
  __syncthreads();
  __threadfence();
}

// ---------- prep kernels ----------
__global__ void cvt_f32_bf16(const float* __restrict__ s, u16* __restrict__ d, int n) {
  int i = blockIdx.x * blockDim.x + threadIdx.x;
  int st = gridDim.x * blockDim.x;
  for (; i < n; i += st) d[i] = f2bf(s[i]);
}
__global__ void bias_sum(const float* __restrict__ a, const float* __restrict__ b,
                         float* __restrict__ o, int n) {
  int i = blockIdx.x * blockDim.x + threadIdx.x;
  if (i < n) o[i] = a[i] + b[i];
}
__global__ void zero_u32(unsigned* __restrict__ p, int n) {
  int i = blockIdx.x * blockDim.x + threadIdx.x;
  if (i < n) p[i] = 0u;
}

// ---------- batched input GEMM: gx[T*B][4096] = A[T*B][K] @ W[4096][K]^T + bias ----------
// One wave computes one 16x16 output tile. Stored bf16, nontemporal (read once later).
__global__ void __launch_bounds__(THREADS) gx_gemm(
    const u16* __restrict__ A, int K,
    const u16* __restrict__ W, const float* __restrict__ bias,
    u16* __restrict__ gx)
{
  const int lane = threadIdx.x & 31;
  const int wave = threadIdx.x >> 5;
  const int tile = blockIdx.x * 8 + wave;   // (T*B/16) * (4096/16) tiles total
  const int mt = tile >> 8;                 // 0..1023
  const int nt = tile & 255;                // 0..255
  const int lm = lane & 15;
  const int lhi = lane >> 4;
  const int n0 = nt * 16;

  const u16* arow = A + (size_t)(mt * 16 + lm) * K;
  const u16* wrow = W + (size_t)(n0 + lm) * K;

  fx8 acc = {};
  for (int k0 = 0; k0 < K; k0 += 32) {
    __builtin_prefetch(arow + k0 + 256, 0, 3);
    __builtin_prefetch(wrow + k0 + 256, 0, 3);
    bfx16 a = load_a_frag(arow, k0, lhi);
    bfx16 b = load_b_frag(wrow, k0, lhi);
    acc = __builtin_amdgcn_wmma_f32_16x16x32_bf16(false, a, false, b, (short)0, acc,
                                                  false, false);
  }
  const float bv = bias[n0 + lm];
  u16* out = gx + (size_t)(mt * 16 + lhi * 8) * G4H + n0 + lm;
#pragma unroll
  for (int v = 0; v < 8; ++v) {
    __builtin_nontemporal_store(f2bf(acc[v] + bv), out + (size_t)v * G4H);
  }
}

// ---------- persistent bidirectional scan for one layer ----------
// Grid: 2 dirs * NBLK blocks. Per timestep:
//   TDM: stage h[32][1024] bf16 (64KB) global -> LDS, s_wait_tensorcnt
//   Phase A: G[dir] = h @ Whh^T (A-fragments from LDS, B from L2-resident weights)
//   barrier
//   Phase B: gates, c/h update, write y (bf16 for next layer, fp32 to d_out on last)
//   barrier
__global__ void __launch_bounds__(THREADS) scan_layer(
    const u16* __restrict__ gxf, const u16* __restrict__ gxr,
    const u16* __restrict__ whhf, const u16* __restrict__ whhr,
    const float* __restrict__ h0, const float* __restrict__ c0,   // [2][B][H] (this layer)
    float* __restrict__ G,          // [2][B][4096]
    u16*   __restrict__ hbuf,       // [2][B][H] bf16
    float* __restrict__ cbuf,       // [2][B][H]
    u16*   __restrict__ yout,       // [T][B][2H] bf16
    float* __restrict__ fout,       // nullptr, or d_out [T][B][2H] f32
    unsigned* __restrict__ bars)    // 2 dirs * {cnt, gen}
{
  __shared__ u16 hsh[BB * HH];      // 64 KB staging tile for h

  const int dir  = blockIdx.x >> 5;   // NBLK == 32
  const int blk  = blockIdx.x & (NBLK - 1);
  const int tid  = threadIdx.x;
  const int lane = tid & 31;
  const int wave = tid >> 5;
  const int lm   = lane & 15;
  const int lhi  = lane >> 4;

  const u16* gx  = dir ? gxr : gxf;
  const u16* whh = dir ? whhr : whhf;
  unsigned* cnt = bars + dir * 2;
  unsigned* gen = cnt + 1;

  u16*   hb = hbuf + (size_t)dir * BB * HH;
  float* cb = cbuf + (size_t)dir * BB * HH;
  float* Gd = G    + (size_t)dir * BB * G4H;

  // Initialize recurrent state from h0/c0 (zeroes in reference, but honor inputs).
  for (int i = blk * THREADS + tid; i < BB * HH; i += NBLK * THREADS) {
    hb[i] = f2bf(h0[(size_t)dir * BB * HH + i]);
    cb[i] = c0[(size_t)dir * BB * HH + i];
  }
  gbar(cnt, gen, NBLK);

  // ---- Tensor DMA descriptor (ISA ch.8): 2D tile, 32 rows x 1024 cols, 2B elems ----
  const unsigned long long ga = (unsigned long long)hb;
  const unsigned lds_addr = (unsigned)(unsigned long long)&hsh[0];  // addr[31:0] = LDS offset
  ui32x4 g0;
  g0[0] = 1u;                                       // count=1, is_restore=0, gather off
  g0[1] = lds_addr;                                 // lds_addr[63:32]
  g0[2] = (unsigned)(ga & 0xFFFFFFFFu);             // global_addr[95:64]
  g0[3] = (unsigned)((ga >> 32) & 0x01FFFFFFu) | 0x80000000u;  // addr[120:96] | type=2
  si32x8 g1;
  g1[0] = (int)(1u << 16);                          // workgroup_mask=0, data_size=1 (2B)
  g1[1] = (int)((unsigned)HH << 16);                // tensor_dim0[15:0]=1024
  g1[2] = (int)((unsigned)BB << 16);                // tensor_dim0 hi=0, tensor_dim1[15:0]=32
  g1[3] = (int)((unsigned)HH << 16);                // tensor_dim1 hi=0, tile_dim0=1024
  g1[4] = (int)BB;                                  // tile_dim1=32, tile_dim2=0
  g1[5] = (int)HH;                                  // tensor_dim0_stride lo = 1024
  g1[6] = 0;                                        // stride0 hi, stride1 lo
  g1[7] = 0;                                        // stride1 hi
  si32x4 gz = {0, 0, 0, 0};                         // groups 2/3 unused (2D)
  si32x8 gz8 = {0, 0, 0, 0, 0, 0, 0, 0};            // trailing group (clang-23 6-arg form)

  const int nt = blk * 8 + wave;        // 0..255 gate-column tile
  const int n0 = nt * 16;
  const u16* wrow  = whh + (size_t)(n0 + lm) * HH;
  const u16* hrow0 = hsh + (size_t)lm * HH;          // M-tile 0: batch rows 0-15
  const u16* hrow1 = hsh + (size_t)(16 + lm) * HH;   // M-tile 1: batch rows 16-31

  for (int t = 0; t < TT; ++t) {
    const int te = dir ? (TT - 1 - t) : t;

    // ---- Stage h into LDS via the Tensor Data Mover ----
    if (wave == 0) {
      __builtin_amdgcn_tensor_load_to_lds(g0, g1, gz, gz, gz8, 0);
      __builtin_amdgcn_s_wait_tensorcnt(0);
    }
    __syncthreads();

    // ---- Phase A: recurrent GEMM slab, K = H = 1024 (32 chained bf16 WMMAs) ----
    fx8 acc0 = {};
    fx8 acc1 = {};
    for (int k0 = 0; k0 < HH; k0 += 32) {
      bfx16 b  = load_b_frag(wrow, k0, lhi);
      bfx16 a0 = load_a_frag(hrow0, k0, lhi);
      bfx16 a1 = load_a_frag(hrow1, k0, lhi);
      acc0 = __builtin_amdgcn_wmma_f32_16x16x32_bf16(false, a0, false, b, (short)0, acc0,
                                                     false, false);
      acc1 = __builtin_amdgcn_wmma_f32_16x16x32_bf16(false, a1, false, b, (short)0, acc1,
                                                     false, false);
    }
    {
      float* g0p = Gd + (size_t)(lhi * 8) * G4H + n0 + lm;
      float* g1p = g0p + (size_t)16 * G4H;
#pragma unroll
      for (int v = 0; v < 8; ++v) {
        g0p[(size_t)v * G4H] = acc0[v];
        g1p[(size_t)v * G4H] = acc1[v];
      }
    }
    gbar(cnt, gen, NBLK);

    // ---- Phase B: hard gates + state update, 4 cells per thread ----
    const u16* gxt = gx + (size_t)te * BB * G4H;
#pragma unroll
    for (int j = 0; j < 4; ++j) {
      const int idx = blk * THREADS + tid + j * (NBLK * THREADS);  // < B*H
      const int r = idx >> 10;          // batch row
      const int c = idx & (HH - 1);     // hidden column
      const float* gr    = Gd  + (size_t)r * G4H;
      const u16*   gxrow = gxt + (size_t)r * G4H;
      float gi = gr[c]            + bf2f(gxrow[c]);
      float gf = gr[HH + c]       + bf2f(gxrow[HH + c]);
      float gg = gr[2 * HH + c]   + bf2f(gxrow[2 * HH + c]);
      float go = gr[3 * HH + c]   + bf2f(gxrow[3 * HH + c]);
      float ig = clampf(SLOPE * gi + OFFS, 0.f, 1.f);
      float fg = clampf(SLOPE * gf + OFFS, 0.f, 1.f);
      float cg = clampf(gg, -1.f, 1.f);
      float og = clampf(SLOPE * go + OFFS, 0.f, 1.f);
      float cn = fg * cb[idx] + ig * cg;
      cb[idx] = cn;
      float hv = og * clampf(cn, -1.f, 1.f);
      hb[idx] = f2bf(hv);
      size_t yi = ((size_t)te * BB + r) * (2 * HH) + (size_t)dir * HH + c;
      yout[yi] = f2bf(hv);
      if (fout) fout[yi] = hv;
    }
    gbar(cnt, gen, NBLK);
  }
}

// ---------- host orchestration ----------
extern "C" void kernel_launch(void* const* d_in, const int* in_sizes, int n_in,
                              void* d_out, int out_size, void* d_ws, size_t ws_size,
                              hipStream_t stream) {
  (void)in_sizes; (void)n_in; (void)out_size; (void)ws_size;

  const float* x  = (const float*)d_in[0];
  const float* h0 = (const float*)d_in[1];
  const float* c0 = (const float*)d_in[2];
  const float *wih[NL][2], *whh[NL][2], *bih[NL][2], *bhh[NL][2];
  for (int l = 0; l < NL; ++l)
    for (int d = 0; d < 2; ++d) {
      int base = 3 + l * 8 + d * 4;
      wih[l][d] = (const float*)d_in[base + 0];
      whh[l][d] = (const float*)d_in[base + 1];
      bih[l][d] = (const float*)d_in[base + 2];
      bhh[l][d] = (const float*)d_in[base + 3];
    }

  // Workspace carve-out
  char* ws = (char*)d_ws;
  size_t off = 0;
  auto take = [&](size_t bytes) -> char* {
    char* p = ws + off;
    off = (off + bytes + 255) & ~(size_t)255;
    return p;
  };

  u16* xin = (u16*)take((size_t)TT * BB * INSZ * 2);
  u16* ybuf[2];
  ybuf[0] = (u16*)take((size_t)TT * BB * 2 * HH * 2);
  ybuf[1] = (u16*)take((size_t)TT * BB * 2 * HH * 2);
  u16* wihb[NL][2];
  u16* whhb[NL][2];
  float* biasb[NL][2];
  for (int l = 0; l < NL; ++l) {
    const int K = (l == 0) ? INSZ : 2 * HH;
    for (int d = 0; d < 2; ++d) {
      wihb[l][d]  = (u16*)take((size_t)G4H * K * 2);
      whhb[l][d]  = (u16*)take((size_t)G4H * HH * 2);
      biasb[l][d] = (float*)take((size_t)G4H * 4);
    }
  }
  u16* gxb[2];
  gxb[0] = (u16*)take((size_t)TT * BB * G4H * 2);
  gxb[1] = (u16*)take((size_t)TT * BB * G4H * 2);
  float* G     = (float*)take((size_t)2 * BB * G4H * 4);
  u16*   hbuf  = (u16*)take((size_t)2 * BB * HH * 2);
  float* cbuf  = (float*)take((size_t)2 * BB * HH * 4);
  unsigned* bars = (unsigned*)take(4096);

  auto cvt_grid = [](long n) -> int {
    long b = (n + THREADS - 1) / THREADS;
    return (int)(b > 8192 ? 8192 : b);
  };

  // Reset barriers; convert inputs/weights to bf16; fuse biases.
  zero_u32<<<4, THREADS, 0, stream>>>(bars, 1024);
  {
    long n = (long)TT * BB * INSZ;
    cvt_f32_bf16<<<cvt_grid(n), THREADS, 0, stream>>>(x, xin, (int)n);
  }
  for (int l = 0; l < NL; ++l) {
    const int K = (l == 0) ? INSZ : 2 * HH;
    for (int d = 0; d < 2; ++d) {
      long nih = (long)G4H * K;
      long nhh = (long)G4H * HH;
      cvt_f32_bf16<<<cvt_grid(nih), THREADS, 0, stream>>>(wih[l][d], wihb[l][d], (int)nih);
      cvt_f32_bf16<<<cvt_grid(nhh), THREADS, 0, stream>>>(whh[l][d], whhb[l][d], (int)nhh);
      bias_sum<<<(G4H + THREADS - 1) / THREADS, THREADS, 0, stream>>>(
          bih[l][d], bhh[l][d], biasb[l][d], G4H);
    }
  }

  // Layer pipeline: parallel gx GEMMs, then the persistent bidirectional scan.
  const int gx_blocks = ((TT * BB / 16) * (G4H / 16)) / 8;  // one wave per 16x16 tile
  for (int l = 0; l < NL; ++l) {
    const u16* Ain = (l == 0) ? xin : ybuf[(l - 1) & 1];
    const int K = (l == 0) ? INSZ : 2 * HH;
    for (int d = 0; d < 2; ++d) {
      gx_gemm<<<gx_blocks, THREADS, 0, stream>>>(Ain, K, wihb[l][d], biasb[l][d], gxb[d]);
    }
    scan_layer<<<2 * NBLK, THREADS, 0, stream>>>(
        gxb[0], gxb[1], whhb[l][0], whhb[l][1],
        h0 + (size_t)2 * l * BB * HH, c0 + (size_t)2 * l * BB * HH,
        G, hbuf, cbuf, ybuf[l & 1],
        (l == NL - 1) ? (float*)d_out : nullptr,
        bars + l * 8);
  }
}
